// TernaryLinear_68573447848569
// MI455X (gfx1250) — compile-verified, matrix operand order checked by reading, not run on hardware
//
#include <hip/hip_runtime.h>
#include <stdint.h>

// ---------------------------------------------------------------------------
// TernaryLinear for MI455X (gfx1250): y = x @ ternary(w)^T
//   M=8192, N=11008, K=4096  -> compute-bound bf16 WMMA GEMM.
//   x split into bf16 hi+lo terms (near-f32 accuracy, ternary w is exact bf16)
// ---------------------------------------------------------------------------

#if defined(__HIP_DEVICE_COMPILE__) && __has_builtin(__builtin_amdgcn_global_load_async_to_lds_b128)
#define HAVE_ASYNC_LDS 1
#else
#define HAVE_ASYNC_LDS 0
#endif

#if HAVE_ASYNC_LDS
#if __has_builtin(__builtin_amdgcn_s_wait_asynccnt)
#define WAIT_ASYNC_LDS() __builtin_amdgcn_s_wait_asynccnt(0)
#else
#define WAIT_ASYNC_LDS() asm volatile("s_wait_asynccnt 0" ::: "memory")
#endif
#else
#define WAIT_ASYNC_LDS() ((void)0)
#endif

typedef float  v8f    __attribute__((ext_vector_type(8)));
typedef __bf16 v16bf  __attribute__((ext_vector_type(16)));
typedef __bf16 bf16x8 __attribute__((ext_vector_type(8)));
typedef __bf16 bf16x4 __attribute__((ext_vector_type(4)));
typedef int    v4i    __attribute__((ext_vector_type(4)));

#if HAVE_ASYNC_LDS
// builtin signature (from clang diagnostic): (v4i addrspace(1)*, v4i addrspace(3)*, imm int, imm int)
typedef __attribute__((address_space(1))) v4i* gptr_v4i;
typedef __attribute__((address_space(3))) v4i* lptr_v4i;
#endif

constexpr int K_DIM = 4096;
constexpr int N_DIM = 11008;
constexpr int M_DIM = 8192;
constexpr long long TOTAL_W = (long long)N_DIM * K_DIM;  // 45088768

constexpr int BM = 128, BN = 128, BK = 32;
constexpr int LDW  = BK + 8;        // 40 bf16 = 80B row stride (16B aligned, bank-staggered)
constexpr int TILE = BM * LDW;      // elems per LDS tile
constexpr int BUFSZ = 3 * TILE;     // A_hi + A_lo + B
constexpr int NK = K_DIM / BK;      // 128 K-steps
constexpr int NB_RED = 2048;

// ---------------------------------------------------------------------------
// Stage 1: deterministic two-stage reduction of sum(|w|)
// ---------------------------------------------------------------------------
__global__ void absmean_partial_kernel(const float* __restrict__ w, float* __restrict__ part) {
  __shared__ float red[256];
  const float4* w4 = (const float4*)w;
  const int total4 = (int)(TOTAL_W / 4);
  float s = 0.0f;
  for (int i = blockIdx.x * 256 + threadIdx.x; i < total4; i += NB_RED * 256) {
    float4 v = w4[i];
    s += fabsf(v.x) + fabsf(v.y) + fabsf(v.z) + fabsf(v.w);
  }
  red[threadIdx.x] = s;
  __syncthreads();
  for (int off = 128; off > 0; off >>= 1) {
    if (threadIdx.x < off) red[threadIdx.x] += red[threadIdx.x + off];
    __syncthreads();
  }
  if (threadIdx.x == 0) part[blockIdx.x] = red[0];
}

__global__ void absmean_final_kernel(const float* __restrict__ part, float* __restrict__ sum) {
  __shared__ float red[256];
  float s = 0.0f;
  for (int i = threadIdx.x; i < NB_RED; i += 256) s += part[i];
  red[threadIdx.x] = s;
  __syncthreads();
  for (int off = 128; off > 0; off >>= 1) {
    if (threadIdx.x < off) red[threadIdx.x] += red[threadIdx.x + off];
    __syncthreads();
  }
  if (threadIdx.x == 0) sum[0] = red[0];
}

// ---------------------------------------------------------------------------
// Stage 2: quantize w -> ternary bf16 (exact in bf16). 90MB -> L2-resident.
// rintf == round-to-nearest-even matches jnp.round.
// ---------------------------------------------------------------------------
__global__ void quantize_w_kernel(const float* __restrict__ w, const float* __restrict__ wsum,
                                  unsigned short* __restrict__ wq_u16) {
  __bf16* wq = (__bf16*)wq_u16;
  const float alpha = wsum[0] * (1.0f / (float)TOTAL_W);
  const float inv = (alpha != 0.0f) ? (1.0f / alpha) : 0.0f;
  const size_t base = ((size_t)blockIdx.x * blockDim.x + threadIdx.x) * 8;
  const float4* p = (const float4*)(w + base);
  float4 a = p[0], b = p[1];
  float fv[8] = {a.x, a.y, a.z, a.w, b.x, b.y, b.z, b.w};
  bf16x8 q;
#pragma unroll
  for (int e = 0; e < 8; ++e) {
    float t = fminf(1.0f, fmaxf(-1.0f, fv[e] * inv));
    q[e] = (__bf16)rintf(t);
  }
  *(bf16x8*)(wq + base) = q;
}

// ---------------------------------------------------------------------------
// Stage 3: bf16 WMMA GEMM, 128x128x32 tiles, double-buffered LDS,
// 8 wave32 waves (2x4), each wave: 4x2 16x16 f32 accumulators,
// 2 WMMA (hi+lo) per accumulator per K-step = 16 v_wmma per wave per step.
// ---------------------------------------------------------------------------
__device__ inline v16bf ld_frag(const __bf16* p) {
  // fragment layout: lanes 0-15 take K[0..7] + K[16..23], lanes 16-31 K[8..15] + K[24..31];
  // caller passes p already offset by (lane>=16 ? 8 : 0)
  bf16x8 c0 = *(const bf16x8*)p;
  bf16x8 c1 = *(const bf16x8*)(p + 16);
  v16bf r;
#pragma unroll
  for (int e = 0; e < 8; ++e) { r[e] = c0[e]; r[e + 8] = c1[e]; }
  return r;
}

template <bool PREQ>
__global__ __launch_bounds__(256) void ternary_gemm_kernel(const float* __restrict__ x,
                                                           const void* __restrict__ wsrc,
                                                           const float* __restrict__ wsum,
                                                           float* __restrict__ out) {
  __shared__ __attribute__((aligned(16))) __bf16 smem[2 * BUFSZ];

  const int tid   = threadIdx.x;
  const int lane  = tid & 31;
  const int lr    = lane & 15;
  const int lh    = lane >> 4;
  const int wave  = tid >> 5;     // 0..7
  const int waveM = wave & 1;     // 2 waves in M
  const int waveN = wave >> 1;    // 4 waves in N
  const int gm0 = blockIdx.y * BM;
  const int gn0 = blockIdx.x * BN;

  float invAlpha = 0.0f;
  if constexpr (!PREQ) {
    const float alpha = wsum[0] * (1.0f / (float)TOTAL_W);
    invAlpha = (alpha != 0.0f) ? (1.0f / alpha) : 0.0f;
  }

  float4 areg[4];
  bf16x8 bregq[2];   (void)bregq;
  float4 bregf[2][2]; (void)bregf;

  // A tile: 128 rows x 32 K f32; 256 threads x 4 float4 (coalesced)
  auto loadA = [&](int kt) {
#pragma unroll
    for (int c = 0; c < 4; ++c) {
      int idx = c * 256 + tid;
      int r = idx >> 3, c4 = idx & 7;
      areg[c] = *(const float4*)(x + (size_t)(gm0 + r) * K_DIM + kt * BK + c4 * 4);
    }
  };

  // split f32 -> bf16 hi + bf16 lo, store both LDS tiles
  auto storeA = [&](int buf) {
    __bf16* sAhi = smem + buf * BUFSZ;
    __bf16* sAlo = sAhi + TILE;
#pragma unroll
    for (int c = 0; c < 4; ++c) {
      int idx = c * 256 + tid;
      int r = idx >> 3, c4 = idx & 7;
      float f[4] = {areg[c].x, areg[c].y, areg[c].z, areg[c].w};
      bf16x4 h, l;
#pragma unroll
      for (int e = 0; e < 4; ++e) {
        __bf16 hb = (__bf16)f[e];
        h[e] = hb;
        l[e] = (__bf16)(f[e] - (float)hb);
      }
      int off = r * LDW + c4 * 4;
      *(bf16x4*)(sAhi + off) = h;
      *(bf16x4*)(sAlo + off) = l;
    }
  };

  // B tile: 128 N-rows x 32 K. PREQ+async: DMA straight into LDS.
  auto loadB_pre = [&](int kt) {
    if constexpr (PREQ) {
#if !HAVE_ASYNC_LDS
#pragma unroll
      for (int c = 0; c < 2; ++c) {
        int idx = c * 256 + tid;
        int r = idx >> 2, c8 = idx & 3;
        bregq[c] = *(const bf16x8*)((const __bf16*)wsrc + (size_t)(gn0 + r) * K_DIM + kt * BK + c8 * 8);
      }
#else
      (void)kt;
#endif
    } else {
#pragma unroll
      for (int c = 0; c < 2; ++c) {
        int idx = c * 256 + tid;
        int r = idx >> 2, c8 = idx & 3;
        const float* p = (const float*)wsrc + (size_t)(gn0 + r) * K_DIM + kt * BK + c8 * 8;
        bregf[c][0] = ((const float4*)p)[0];
        bregf[c][1] = ((const float4*)p)[1];
      }
    }
  };

  auto loadB_post = [&](int buf, int kt) {
    __bf16* sB = smem + buf * BUFSZ + 2 * TILE;
    if constexpr (PREQ) {
#if HAVE_ASYNC_LDS
#pragma unroll
      for (int c = 0; c < 2; ++c) {
        int idx = c * 256 + tid;
        int r = idx >> 2, c8 = idx & 3;
        const __bf16* g = (const __bf16*)wsrc + (size_t)(gn0 + r) * K_DIM + kt * BK + c8 * 8;
        __builtin_amdgcn_global_load_async_to_lds_b128(
            (gptr_v4i)(uintptr_t)g,
            (lptr_v4i)(uint32_t)(uintptr_t)(sB + r * LDW + c8 * 8),
            0, 0);
      }
#else
      (void)kt;
#pragma unroll
      for (int c = 0; c < 2; ++c) {
        int idx = c * 256 + tid;
        int r = idx >> 2, c8 = idx & 3;
        *(bf16x8*)(sB + r * LDW + c8 * 8) = bregq[c];
      }
#endif
    } else {
      (void)kt;
#pragma unroll
      for (int c = 0; c < 2; ++c) {
        int idx = c * 256 + tid;
        int r = idx >> 2, c8 = idx & 3;
        float fv[8] = {bregf[c][0].x, bregf[c][0].y, bregf[c][0].z, bregf[c][0].w,
                       bregf[c][1].x, bregf[c][1].y, bregf[c][1].z, bregf[c][1].w};
        bf16x8 q;
#pragma unroll
        for (int e = 0; e < 8; ++e) {
          float t = fminf(1.0f, fmaxf(-1.0f, fv[e] * invAlpha));
          q[e] = (__bf16)rintf(t);
        }
        *(bf16x8*)(sB + r * LDW + c8 * 8) = q;
      }
    }
  };

  v8f acc[4][2] = {};

  auto compute = [&](int buf) {
    const __bf16* sAhi = smem + buf * BUFSZ;
    const __bf16* sAlo = sAhi + TILE;
    const __bf16* sB   = sAlo + TILE;
    const int kof = lh * 8;
    v16bf bfrag[2];
#pragma unroll
    for (int j = 0; j < 2; ++j)
      bfrag[j] = ld_frag(sB + (waveN * 32 + j * 16 + lr) * LDW + kof);
#pragma unroll
    for (int i = 0; i < 4; ++i) {
      const int rowOff = (waveM * 64 + i * 16 + lr) * LDW + kof;
      v16bf ah = ld_frag(sAhi + rowOff);
      v16bf al = ld_frag(sAlo + rowOff);
#pragma unroll
      for (int j = 0; j < 2; ++j) {
        acc[i][j] = __builtin_amdgcn_wmma_f32_16x16x32_bf16(false, ah, false, bfrag[j],
                                                            (short)0, acc[i][j], false, false);
        acc[i][j] = __builtin_amdgcn_wmma_f32_16x16x32_bf16(false, al, false, bfrag[j],
                                                            (short)0, acc[i][j], false, false);
      }
    }
  };

  // ---- pipelined main loop (double-buffered LDS) ----
  loadA(0);
  loadB_pre(0);
  storeA(0);
  loadB_post(0, 0);

#pragma unroll 1
  for (int kt = 0; kt < NK; ++kt) {
    const int cur = kt & 1;
    const bool hasNext = (kt + 1) < NK;
    if (hasNext) { loadA(kt + 1); loadB_pre(kt + 1); }
    WAIT_ASYNC_LDS();        // async B writes for 'cur' complete
    __syncthreads();         // all waves' LDS fills for 'cur' visible; 'cur^1' reads retired
    if (hasNext) { storeA(cur ^ 1); loadB_post(cur ^ 1, kt + 1); }
    compute(cur);
  }

  // ---- epilogue: C layout VGPR v: M = base + (lane>=16?8:0) + v, N = base + lane%16 ----
#pragma unroll
  for (int i = 0; i < 4; ++i) {
#pragma unroll
    for (int j = 0; j < 2; ++j) {
      const int rbase = gm0 + waveM * 64 + i * 16 + lh * 8;
      const int cbase = gn0 + waveN * 32 + j * 16 + lr;
#pragma unroll
      for (int v = 0; v < 8; ++v)
        out[(size_t)(rbase + v) * N_DIM + cbase] = acc[i][j][v];
    }
  }
}

// ---------------------------------------------------------------------------
extern "C" void kernel_launch(void* const* d_in, const int* in_sizes, int n_in,
                              void* d_out, int out_size, void* d_ws, size_t ws_size,
                              hipStream_t stream) {
  (void)in_sizes; (void)n_in; (void)out_size;
  const float* x = (const float*)d_in[0];
  const float* w = (const float*)d_in[1];
  float* partials = (float*)d_ws;                                 // [0 .. 2048)
  float* wsum = partials + NB_RED;                                // [2048]
  unsigned short* wq = (unsigned short*)((char*)d_ws + 16384);    // bf16 weight, 90MB
  float* out = (float*)d_out;

  absmean_partial_kernel<<<NB_RED, 256, 0, stream>>>(w, partials);
  absmean_final_kernel<<<1, 256, 0, stream>>>(partials, wsum);

  dim3 grid(N_DIM / BN, M_DIM / BM);  // 86 x 64, exact
  const size_t need = 16384 + (size_t)TOTAL_W * 2;
  if (ws_size >= need) {
    quantize_w_kernel<<<(int)(TOTAL_W / 8 / 256), 256, 0, stream>>>(w, wsum, wq);
    ternary_gemm_kernel<true><<<grid, 256, 0, stream>>>(x, (const void*)wq, wsum, out);
  } else {
    // workspace too small: fuse quantization into the B-tile load (f32 weight reads)
    ternary_gemm_kernel<false><<<grid, 256, 0, stream>>>(x, (const void*)w, wsum, out);
  }
}
